// EncoderBlock_75153337745679
// MI455X (gfx1250) — compile-verified, matrix operand order checked by reading, not run on hardware
//
#include <hip/hip_runtime.h>

typedef _Float16 half_t;
typedef __attribute__((ext_vector_type(16))) _Float16 v16h;
typedef __attribute__((ext_vector_type(8)))  _Float16 v8h;
typedef __attribute__((ext_vector_type(4)))  _Float16 v4h;
typedef __attribute__((ext_vector_type(8)))  float    v8f;
typedef __attribute__((ext_vector_type(4)))  float    v4f;

__device__ inline v8f v8f_zero() {
  v8f v;
#pragma unroll
  for (int i = 0; i < 8; ++i) v[i] = 0.0f;
  return v;
}

__device__ inline v8f wmma_f16(v16h a, v16h b, v8f c) {
  return __builtin_amdgcn_wmma_f32_16x16x32_f16(
      false, a, false, b, (short)0, c, false, false);
}

// Load a 16x32 f16 WMMA A/B fragment from a row-major tile (row stride in
// halves). Per ISA 7.12.2: lane (l&15) = row (A) / col (B); K mapping is
// k = 16*(v>>2) + 8*(l>>4) + 2*(v&3) + {0,1}  ->  two contiguous 16B reads.
__device__ inline v16h load_frag16(const half_t* p0, int strideH) {
  const int lane = threadIdx.x & 31;
  const half_t* p = p0 + (lane & 15) * strideH + (lane >> 4) * 8;
  v8h a = *(const v8h*)p;
  v8h b = *(const v8h*)(p + 16);
  v16h o;
#pragma unroll
  for (int i = 0; i < 8; ++i) { o[i] = a[i]; o[i + 8] = b[i]; }
  return o;
}

// ---------------------------------------------------------------------------
// LayerNorm: one row (D=1024) per 256-thread block, f32 in, f16 out.
// ---------------------------------------------------------------------------
__global__ __launch_bounds__(256) void ln_kernel(
    const float* __restrict__ x, const float* __restrict__ g,
    const float* __restrict__ b, half_t* __restrict__ out, int D) {
  const int row = blockIdx.x;
  const int tid = threadIdx.x;
  const float* xr = x + (size_t)row * D;
  v4f v = *(const v4f*)(xr + tid * 4);
  float s = v[0] + v[1] + v[2] + v[3];
  float ss = v[0] * v[0] + v[1] * v[1] + v[2] * v[2] + v[3] * v[3];
#pragma unroll
  for (int off = 1; off < 32; off <<= 1) {
    s += __shfl_xor(s, off, 32);
    ss += __shfl_xor(ss, off, 32);
  }
  __shared__ float sbuf[8], ssbuf[8];
  if ((tid & 31) == 0) { sbuf[tid >> 5] = s; ssbuf[tid >> 5] = ss; }
  __syncthreads();
  float ts = 0.f, tss = 0.f;
#pragma unroll
  for (int i = 0; i < 8; ++i) { ts += sbuf[i]; tss += ssbuf[i]; }
  const float mu = ts / (float)D;
  const float var = tss / (float)D - mu * mu;
  const float rstd = rsqrtf(var + 1e-5f);
  v4f gv = *(const v4f*)(g + tid * 4);
  v4f bv = *(const v4f*)(b + tid * 4);
  v4h ov;
#pragma unroll
  for (int i = 0; i < 4; ++i)
    ov[i] = (half_t)((v[i] - mu) * rstd * gv[i] + bv[i]);
  *(v4h*)(out + (size_t)row * D + tid * 4) = ov;
}

// ---------------------------------------------------------------------------
// WMMA GEMM: out[M,N] = A[M,K](f16) @ W[K,N](f32->f16) + bias (+relu | +res)
// 128x128 block tile, 8 waves (4M x 2N), 32-wide K slabs, double-buffered LDS
// (global loads of slab k+1 issued before the WMMAs of slab k).
// ---------------------------------------------------------------------------
#define GT_M 128
#define GT_N 128
#define GT_K 32
#define GLDS 40  // LDS row stride in halves (16B-aligned, bank-skewed)

template <bool RELU, bool F32OUT>
__global__ __launch_bounds__(256) void gemm_kernel(
    const half_t* __restrict__ A, const float* __restrict__ W,
    const float* __restrict__ bias, const float* __restrict__ residual,
    half_t* __restrict__ outH, float* __restrict__ outF,
    int M, int N, int K) {
  __shared__ __align__(16) half_t As[2][GT_M * GLDS];
  __shared__ __align__(16) half_t Bs[2][GT_N * GLDS];  // W tile transposed [n][k]

  const int tid = threadIdx.x;
  const int wave = tid >> 5;
  const int lane = tid & 31;
  const int bm = blockIdx.x * GT_M;
  const int bn = blockIdx.y * GT_N;
  const int wm = (wave & 3) * 32;   // 2 M-subtiles
  const int wn = (wave >> 2) * 64;  // 4 N-subtiles

  const int arow = tid >> 1;        // 0..127
  const int acol = (tid & 1) * 16;  // 0 or 16
  const int wn4 = (tid & 31) * 4;   // 0..124
  const int wk0 = tid >> 5;         // 0..7

  v8f acc[2][4];
#pragma unroll
  for (int i = 0; i < 2; ++i)
#pragma unroll
    for (int j = 0; j < 4; ++j) acc[i][j] = v8f_zero();

  v8h a0, a1;   // A slab staging regs
  v4f wv[4];    // W slab staging regs

  auto load_slab = [&](int k0) {
    const half_t* src = A + (size_t)(bm + arow) * K + k0 + acol;
    a0 = *(const v8h*)src;
    a1 = *(const v8h*)(src + 8);
#pragma unroll
    for (int kk = 0; kk < 4; ++kk)
      wv[kk] = *(const v4f*)(W + (size_t)(k0 + wk0 + kk * 8) * N + bn + wn4);
  };
  auto stage_slab = [&](int buf) {
    *(v8h*)&As[buf][arow * GLDS + acol] = a0;
    *(v8h*)&As[buf][arow * GLDS + acol + 8] = a1;
#pragma unroll
    for (int kk = 0; kk < 4; ++kk)
#pragma unroll
      for (int i = 0; i < 4; ++i)
        Bs[buf][(wn4 + i) * GLDS + wk0 + kk * 8] = (half_t)wv[kk][i];
  };

  load_slab(0);
  stage_slab(0);
  __syncthreads();

  const int nSlab = K / GT_K;
  for (int s = 0; s < nSlab; ++s) {
    const int cur = s & 1;
    const bool more = (s + 1) < nSlab;
    if (more) load_slab((s + 1) * GT_K);  // overlap with WMMAs below

    v16h af[2], bf[4];
#pragma unroll
    for (int i = 0; i < 2; ++i)
      af[i] = load_frag16(&As[cur][(wm + i * 16) * GLDS], GLDS);
#pragma unroll
    for (int j = 0; j < 4; ++j)
      bf[j] = load_frag16(&Bs[cur][(wn + j * 16) * GLDS], GLDS);
#pragma unroll
    for (int i = 0; i < 2; ++i)
#pragma unroll
      for (int j = 0; j < 4; ++j)
        acc[i][j] = wmma_f16(af[i], bf[j], acc[i][j]);

    if (more) stage_slab(cur ^ 1);
    __syncthreads();  // one barrier/slab: covers nxt-writes and cur-reads
  }

  // Epilogue. C/D layout: vgpr r, lane l -> row = r + 8*(l>>4), col = l&15.
  const int r16 = lane & 15, hiL = lane >> 4;
#pragma unroll
  for (int j = 0; j < 4; ++j) {
    const int n = bn + wn + j * 16 + r16;
    const float bval = bias[n];
#pragma unroll
    for (int i = 0; i < 2; ++i) {
      const size_t base = (size_t)(bm + wm + i * 16 + 8 * hiL) * N + n;
#pragma unroll
      for (int r = 0; r < 8; ++r) {
        float v = acc[i][j][r] + bval;
        if (RELU) v = fmaxf(v, 0.0f);
        const size_t idx = base + (size_t)r * N;
        if (F32OUT) outF[idx] = v + residual[idx];
        else        outH[idx] = (half_t)v;
      }
    }
  }
}

// ---------------------------------------------------------------------------
// Flash attention: grid (S/128, H, B), 8 waves x 16 query rows each.
// Q/K frags straight from global (contiguous 16B per lane); V transposed in
// LDS; P transposed through per-wave LDS; online softmax in fp32.
// ---------------------------------------------------------------------------
__global__ __launch_bounds__(256) void attn_kernel(
    const half_t* __restrict__ Q, const half_t* __restrict__ K,
    const half_t* __restrict__ V, const int* __restrict__ mask,
    half_t* __restrict__ O, int S) {
  const int h = blockIdx.y;
  const int bi = blockIdx.z;
  const int m0 = blockIdx.x * 128;
  const int tid = threadIdx.x, wave = tid >> 5, lane = tid & 31;
  const int r16 = lane & 15, hiL = lane >> 4;
  const int LD = 1024;  // H * d_head

  const half_t* qb = Q + (size_t)bi * S * LD + h * 64;
  const half_t* kb = K + (size_t)bi * S * LD + h * 64;
  const half_t* vb = V + (size_t)bi * S * LD + h * 64;
  const int* mrow = mask + (size_t)bi * S;

  __shared__ __align__(16) half_t Vt[64 * 72];     // [d][key], stride 72
  __shared__ __align__(16) half_t Pb[8][16 * 72];  // per-wave P [row][key]

  const int mw = m0 + wave * 16;
  v16h qf[2];
#pragma unroll
  for (int f = 0; f < 2; ++f)
    qf[f] = load_frag16(qb + (size_t)mw * LD + f * 32, LD);

  v8f oacc[4];
#pragma unroll
  for (int i = 0; i < 4; ++i) oacc[i] = v8f_zero();
  float mst[8], lst[8];
#pragma unroll
  for (int r = 0; r < 8; ++r) { mst[r] = -1e30f; lst[r] = 0.0f; }

  for (int kb0 = 0; kb0 < S; kb0 += 64) {
    {  // cooperative: V block (64 keys x 64 d) -> Vt transposed
      const int key = tid >> 2;
      const int d0 = (tid & 3) * 16;
      const half_t* src = vb + (size_t)(kb0 + key) * LD + d0;
      v8h a = *(const v8h*)src;
      v8h c = *(const v8h*)(src + 8);
#pragma unroll
      for (int i = 0; i < 8; ++i) {
        Vt[(d0 + i) * 72 + key] = a[i];
        Vt[(d0 + 8 + i) * 72 + key] = c[i];
      }
    }
    __syncthreads();

    // scores: 4 key tiles of 16, contraction d_head=64 -> 2 WMMAs each
    v8f sacc[4];
#pragma unroll
    for (int nt = 0; nt < 4; ++nt) {
      const int keyn = kb0 + nt * 16;
      v16h bk0 = load_frag16(kb + (size_t)keyn * LD + 0, LD);
      v16h bk1 = load_frag16(kb + (size_t)keyn * LD + 32, LD);
      v8f sc = v8f_zero();
      sc = wmma_f16(qf[0], bk0, sc);
      sc = wmma_f16(qf[1], bk1, sc);
      const int mk = mrow[keyn + r16];
#pragma unroll
      for (int r = 0; r < 8; ++r)
        sc[r] = (mk == 0) ? -1e30f : sc[r] * 0.125f;  // 1/sqrt(64)
      sacc[nt] = sc;
    }

    // online softmax: row max across 4 tiles + 16 lanes sharing the row
    float nm[8], rsum[8];
#pragma unroll
    for (int r = 0; r < 8; ++r) {
      float v = fmaxf(fmaxf(sacc[0][r], sacc[1][r]),
                      fmaxf(sacc[2][r], sacc[3][r]));
#pragma unroll
      for (int off = 1; off < 16; off <<= 1)
        v = fmaxf(v, __shfl_xor(v, off, 32));
      nm[r] = fmaxf(mst[r], v);
      const float f = __expf(mst[r] - nm[r]);
      mst[r] = nm[r];
      lst[r] *= f;
#pragma unroll
      for (int nt = 0; nt < 4; ++nt) oacc[nt][r] *= f;
      rsum[r] = 0.0f;
    }
#pragma unroll
    for (int nt = 0; nt < 4; ++nt) {
#pragma unroll
      for (int r = 0; r < 8; ++r) {
        const float p = __expf(sacc[nt][r] - mst[r]);
        rsum[r] += p;
        Pb[wave][(r + 8 * hiL) * 72 + nt * 16 + r16] = (half_t)p;
      }
    }
#pragma unroll
    for (int r = 0; r < 8; ++r) {
      float v = rsum[r];
#pragma unroll
      for (int off = 1; off < 16; off <<= 1) v += __shfl_xor(v, off, 32);
      lst[r] += v;
    }

    // P @ V  (wave-private Pb: per-wave LDS ordering is in-order, no barrier)
    v16h pf0 = load_frag16(&Pb[wave][0], 72);
    v16h pf1 = load_frag16(&Pb[wave][32], 72);
#pragma unroll
    for (int nt = 0; nt < 4; ++nt) {
      v16h bv0 = load_frag16(&Vt[(nt * 16) * 72 + 0], 72);
      v16h bv1 = load_frag16(&Vt[(nt * 16) * 72 + 32], 72);
      oacc[nt] = wmma_f16(pf0, bv0, oacc[nt]);
      oacc[nt] = wmma_f16(pf1, bv1, oacc[nt]);
    }
    __syncthreads();  // protect Vt before next block overwrites it
  }

  half_t* ob = O + (size_t)bi * S * LD + h * 64;
#pragma unroll
  for (int r = 0; r < 8; ++r) {
    const float inv = 1.0f / lst[r];
    const int m = mw + r + 8 * hiL;
#pragma unroll
    for (int nt = 0; nt < 4; ++nt)
      ob[(size_t)m * LD + nt * 16 + r16] = (half_t)(oacc[nt][r] * inv);
  }
}

// ---------------------------------------------------------------------------
extern "C" void kernel_launch(void* const* d_in, const int* in_sizes, int n_in,
                              void* d_out, int out_size, void* d_ws,
                              size_t ws_size, hipStream_t stream) {
  const float* x    = (const float*)d_in[0];
  const int*   mask = (const int*)d_in[1];
  const float* wq = (const float*)d_in[2];  const float* bq = (const float*)d_in[3];
  const float* wk = (const float*)d_in[4];  const float* bk = (const float*)d_in[5];
  const float* wv = (const float*)d_in[6];  const float* bv = (const float*)d_in[7];
  const float* wo = (const float*)d_in[8];  const float* bo = (const float*)d_in[9];
  const float* ln1g = (const float*)d_in[10]; const float* ln1b = (const float*)d_in[11];
  const float* ln2g = (const float*)d_in[12]; const float* ln2b = (const float*)d_in[13];
  const float* w1 = (const float*)d_in[14]; const float* b1 = (const float*)d_in[15];
  const float* w2 = (const float*)d_in[16]; const float* b2 = (const float*)d_in[17];

  const int Bn = 2, S = 2048, D = 1024, FF = 4096;
  const int M = Bn * S;  // 4096

  char* ws = (char*)d_ws;
  half_t* h16   = (half_t*)(ws + ((size_t)0 << 20));   //  8 MB
  half_t* q16   = (half_t*)(ws + ((size_t)8 << 20));   //  8 MB
  half_t* k16   = (half_t*)(ws + ((size_t)16 << 20));  //  8 MB
  half_t* v16   = (half_t*)(ws + ((size_t)24 << 20));  //  8 MB
  half_t* ctx16 = (half_t*)(ws + ((size_t)32 << 20));  //  8 MB
  half_t* h2    = (half_t*)(ws + ((size_t)40 << 20));  //  8 MB
  half_t* ff1   = (half_t*)(ws + ((size_t)48 << 20));  // 32 MB
  float*  x1    = (float*)(ws + ((size_t)80 << 20));   // 16 MB

  // 1. LN1
  ln_kernel<<<M, 256, 0, stream>>>(x, ln1g, ln1b, h16, D);

  // 2. Q/K/V projections (f16 out)
  dim3 gqkv(M / 128, D / 128);
  gemm_kernel<false, false><<<gqkv, 256, 0, stream>>>(
      h16, wq, bq, nullptr, q16, nullptr, M, D, D);
  gemm_kernel<false, false><<<gqkv, 256, 0, stream>>>(
      h16, wk, bk, nullptr, k16, nullptr, M, D, D);
  gemm_kernel<false, false><<<gqkv, 256, 0, stream>>>(
      h16, wv, bv, nullptr, v16, nullptr, M, D, D);

  // 3. attention
  attn_kernel<<<dim3(S / 128, 16, Bn), 256, 0, stream>>>(q16, k16, v16, mask,
                                                         ctx16, S);

  // 4. output projection + residual (f32 out)
  gemm_kernel<false, true><<<gqkv, 256, 0, stream>>>(
      ctx16, wo, bo, x, nullptr, x1, M, D, D);

  // 5. LN2
  ln_kernel<<<M, 256, 0, stream>>>(x1, ln2g, ln2b, h2, D);

  // 6. FF1 + relu (f16 out)
  dim3 gff1(M / 128, FF / 128);
  gemm_kernel<true, false><<<gff1, 256, 0, stream>>>(
      h2, w1, b1, nullptr, ff1, nullptr, M, FF, D);

  // 7. FF2 + residual -> output (fp32)
  dim3 gff2(M / 128, D / 128);
  gemm_kernel<false, true><<<gff2, 256, 0, stream>>>(
      ff1, w2, b2, x1, nullptr, (float*)d_out, M, D, FF);
}
